// AdditiveAttention_79328045957428
// MI455X (gfx1250) — matrix operand order repaired and checked by reading, not compiled, not run on hardware
//
#include <hip/hip_runtime.h>
#include <hip/hip_bf16.h>
#include <math.h>
#include <stdint.h>

// Problem constants (from reference): B=32, S=4096, H=512, A=256
#define B_ 32
#define S_ 4096
#define H_ 512
#define A_ 256

typedef __attribute__((ext_vector_type(16))) __bf16 v16bf;
typedef __attribute__((ext_vector_type(8)))  __bf16 v8bf;
typedef __attribute__((ext_vector_type(8)))  float  v8f;
typedef __attribute__((ext_vector_type(4)))  float  v4f;

// LDS layout constants (padded for bank-conflict-free ds_load_b128)
#define A_ROW_STRIDE 144        // 32 f32 (128B) + 16B pad ; 16B aligned
#define A_SLAB       (32 * A_ROW_STRIDE)        // 4608B per strip per step
#define B_COL_STRIDE 80         // 32 bf16 (64B) + 16B pad ; 16B aligned
#define B_SLAB       (256 * B_COL_STRIDE)       // 20480B per step

// ---------------------------------------------------------------------------
// Hardware tanh (V_TANH_F32 trans unit on gfx1250)
// ---------------------------------------------------------------------------
__device__ __forceinline__ float fast_tanh(float x) {
#if __has_builtin(__builtin_amdgcn_tanhf)
    return __builtin_amdgcn_tanhf(x);
#elif __has_builtin(__builtin_amdgcn_tanh_f32)
    return __builtin_amdgcn_tanh_f32(x);
#else
    float r;
    asm volatile("v_tanh_f32 %0, %1\n\tv_nop" : "=v"(r) : "v"(x));
    return r;
#endif
}

// ---------------------------------------------------------------------------
// CDNA5 async global->LDS copy (ASYNCcnt domain). Each lane moves 16 bytes.
// ---------------------------------------------------------------------------
__device__ __forceinline__ void async_b128(uint32_t lds_addr, const void* gaddr) {
    asm volatile("global_load_async_to_lds_b128 %0, %1, off"
                 :: "v"(lds_addr), "v"(gaddr) : "memory");
}
#define WAIT_ASYNC_12() asm volatile("s_wait_asynccnt 0xc" ::: "memory")
#define WAIT_ASYNC_0()  asm volatile("s_wait_asynccnt 0x0" ::: "memory")

// 16B-aligned LDS vector loads (lower to ds_load_b128 pairs)
__device__ __forceinline__ v8f load_v8f(const char* p) {
    v4f lo = *(const v4f*)p;
    v4f hi = *(const v4f*)(p + 16);
    return __builtin_shufflevector(lo, hi, 0, 1, 2, 3, 4, 5, 6, 7);
}
__device__ __forceinline__ v16bf load_v16bf(const char* p) {
    v8bf lo = *(const v8bf*)p;
    v8bf hi = *(const v8bf*)(p + 16);
    return __builtin_shufflevector(lo, hi, 0, 1, 2, 3, 4, 5, 6, 7,
                                           8, 9, 10, 11, 12, 13, 14, 15);
}

// ---------------------------------------------------------------------------
// Kernel 1: W_enc [H][A] f32 -> W_T [A][H] bf16 (K-major)
// ---------------------------------------------------------------------------
__global__ void wt_convert_kernel(const float* __restrict__ Wenc,
                                  __bf16* __restrict__ WT) {
    int idx = blockIdx.x * blockDim.x + threadIdx.x;   // H*A = 131072 threads
    int a = idx & (A_ - 1);
    int h = idx >> 8;                                  // coalesced read of Wenc
    WT[(size_t)a * H_ + h] = (__bf16)Wenc[idx];
}

// ---------------------------------------------------------------------------
// Kernel 2: dp[b][a] = decoder_hidden[b].W_dec[:,a] + b_dec[a] + b_enc[a]
// (b_e omitted: softmax is shift-invariant -> cannot affect attn/context)
// ---------------------------------------------------------------------------
__global__ void dec_proj_kernel(const float* __restrict__ dh,
                                const float* __restrict__ Wdec,
                                const float* __restrict__ benc,
                                const float* __restrict__ bdec,
                                float* __restrict__ dp) {
    int idx = blockIdx.x * blockDim.x + threadIdx.x;   // B*A = 8192 threads
    int b = idx >> 8;
    int a = idx & (A_ - 1);
    float s = bdec[a] + benc[a];
    const float* h = dh + (size_t)b * H_;
    #pragma unroll 8
    for (int k = 0; k < H_; ++k) s += h[k] * Wdec[(size_t)k * A_ + a];
    dp[idx] = s;
}

// ---------------------------------------------------------------------------
// One K-step consume: 2 A-frags (rows 0..15 / 16..31 of strip) x 8 B-tiles.
// All operands come from LDS (bank-conflict-free padded layouts).
// ---------------------------------------------------------------------------
__device__ __forceinline__ void consume_step(
    const char* __restrict__ aslab, const char* __restrict__ bslab,
    int l15, int hi32, int nh_off, v8f acc0[8], v8f acc1[8])
{
    v8f a0lo = load_v8f(aslab + l15 * A_ROW_STRIDE + 2 * hi32);
    v8f a0hi = load_v8f(aslab + l15 * A_ROW_STRIDE + 32 + 2 * hi32);
    v8f a1lo = load_v8f(aslab + (16 + l15) * A_ROW_STRIDE + 2 * hi32);
    v8f a1hi = load_v8f(aslab + (16 + l15) * A_ROW_STRIDE + 32 + 2 * hi32);
    v16bf af0, af1;
    #pragma unroll
    for (int i = 0; i < 8; ++i) {
        af0[i] = (__bf16)a0lo[i]; af0[8 + i] = (__bf16)a0hi[i];
        af1[i] = (__bf16)a1lo[i]; af1[8 + i] = (__bf16)a1hi[i];
    }
    #pragma unroll
    for (int n = 0; n < 8; ++n) {
        const int col = nh_off + n * 16 + l15;
        v16bf bf = load_v16bf(bslab + col * B_COL_STRIDE + hi32);
        acc0[n] = __builtin_amdgcn_wmma_f32_16x16x32_bf16(
            false, af0, false, bf, (short)0, acc0[n], false, false);
        acc1[n] = __builtin_amdgcn_wmma_f32_16x16x32_bf16(
            false, af1, false, bf, (short)0, acc1[n], false, false);
    }
}

// ---------------------------------------------------------------------------
// Kernel 3 (main): block = 4 waves = 2 S-strips (32 rows) x 2 N-halves (128).
// Per K-step the block cooperatively async-stages the B slab (shared) and the
// two A strip-slabs into a 3-buffer LDS ring (ASYNCcnt, depth 2), then each
// wave runs 16 v_wmma_f32_16x16x32_bf16 fed entirely from LDS.
// ---------------------------------------------------------------------------
__global__ __launch_bounds__(128) void fused_score_kernel(
    const float*  __restrict__ enc,     // [B,S,H] f32
    const __bf16* __restrict__ WT,      // [A,H]  bf16
    const float*  __restrict__ dp,      // [B,A]  dec proj + b_enc
    const float*  __restrict__ We,      // [A]
    float* __restrict__ scores)         // [B,S]  (== attn region of d_out)
{
    __shared__ __align__(16) char ldsB[3][B_SLAB];      // 61440 B
    __shared__ __align__(16) char ldsA[3][2][A_SLAB];   // 27648 B

    const int lane  = threadIdx.x & 31;
    const int wave  = threadIdx.x >> 5;
    const int strip = wave >> 1;              // 0/1 : which 32-row S strip
    const int nh    = wave & 1;               // 0/1 : which 128-col A half
    const int b     = blockIdx.x >> 6;        // 64 blocks per batch (S/64)
    const int sblk  = (blockIdx.x & 63) << 6; // 64 S rows per block
    const int sbase = sblk + strip * 32;

    const int l15  = lane & 15;
    const int hi32 = (lane >> 4) * 32;
    const int nh_off = nh * 128;

    // ---- async A addressing: this wave stages rows nh*16..nh*16+15 of its
    // strip; instr j copies rows arow0+j*4, 16B chunk (lane&7) of each row.
    const int arow0 = nh * 16 + (lane >> 3);
    const char* gaA0 = (const char*)enc
        + ((size_t)b * S_ + sbase + arow0) * H_ * 4 + (size_t)(lane & 7) * 16;
    const uint32_t laA0 = (uint32_t)(uintptr_t)&ldsA[0][strip][0]
        + (uint32_t)(arow0 * A_ROW_STRIDE + (lane & 7) * 16);

    // ---- async B addressing: wave w stages cols w*64..w*64+63; instr j
    // copies cols c0+j*8, 16B chunk (lane&3) of each 64B col-slab.
    const int c0 = wave * 64 + (lane >> 2);
    const char* gaB0 = (const char*)WT + (size_t)c0 * H_ * 2 + (size_t)(lane & 3) * 16;
    const uint32_t laB0 = (uint32_t)(uintptr_t)&ldsB[0][0]
        + (uint32_t)(c0 * B_COL_STRIDE + (lane & 3) * 16);

    v8f acc0[8], acc1[8];
    #pragma unroll
    for (int n = 0; n < 8; ++n)
        #pragma unroll
        for (int r = 0; r < 8; ++r) { acc0[n][r] = 0.0f; acc1[n][r] = 0.0f; }

    // issue slab-set kk (12 async ops per wave: 4 A + 8 B)
    auto issue = [&](int kk) {
        const int buf = kk % 3;
        const uint32_t laA = laA0 + (uint32_t)buf * (2 * A_SLAB);
        const char*    gaA = gaA0 + (size_t)kk * 128;
        #pragma unroll
        for (int j = 0; j < 4; ++j)
            async_b128(laA + j * (4 * A_ROW_STRIDE), gaA + (size_t)j * (4 * H_ * 4));
        const uint32_t laB = laB0 + (uint32_t)buf * B_SLAB;
        const char*    gaB = gaB0 + (size_t)kk * 64;
        #pragma unroll
        for (int j = 0; j < 8; ++j)
            async_b128(laB + j * (8 * B_COL_STRIDE), gaB + (size_t)j * (8 * H_ * 2));
    };

    // ---- prologue: slab-sets 0 and 1 in flight (24 ops)
    issue(0);
    issue(1);

    // ---- steady state: wait own 12-op slab-set kk retired (<=12 outstanding
    // == exactly slab kk+1), publish via barrier, issue kk+2, consume kk.
    #pragma unroll 1
    for (int kk = 0; kk < 14; ++kk) {
        WAIT_ASYNC_12();
        __syncthreads();
        issue(kk + 2);
        consume_step(&ldsA[kk % 3][strip][0], &ldsB[kk % 3][0],
                     l15, hi32, nh_off, acc0, acc1);
    }
    // kk = 14: slabs {14,15} pending; <=12 retires the 12 oldest == slab 14
    WAIT_ASYNC_12();
    __syncthreads();
    consume_step(&ldsA[14 % 3][strip][0], &ldsB[14 % 3][0],
                 l15, hi32, nh_off, acc0, acc1);
    // kk = 15: drain
    WAIT_ASYNC_0();
    __syncthreads();
    consume_step(&ldsA[15 % 3][strip][0], &ldsB[15 % 3][0],
                 l15, hi32, nh_off, acc0, acc1);

    // ---- Epilogue: tanh + dot with W_e over this wave's 128 cols, 16-lane
    // reduction, then cross-wave (N-half) combine through LDS.
    float dpv[8], wev[8], p0[8], p1[8];
    #pragma unroll
    for (int n = 0; n < 8; ++n) {
        const int a = nh_off + n * 16 + l15;
        dpv[n] = dp[(size_t)b * A_ + a];
        wev[n] = We[a];
        p0[n] = 0.0f; p1[n] = 0.0f;   // unused; row partials are q0/q1 below
    }
    float q0[8], q1[8];
    #pragma unroll
    for (int r = 0; r < 8; ++r) { q0[r] = 0.0f; q1[r] = 0.0f; }
    #pragma unroll
    for (int n = 0; n < 8; ++n)
        #pragma unroll
        for (int r = 0; r < 8; ++r) {
            q0[r] = fmaf(fast_tanh(acc0[n][r] + dpv[n]), wev[n], q0[r]);
            q1[r] = fmaf(fast_tanh(acc1[n][r] + dpv[n]), wev[n], q1[r]);
        }
    #pragma unroll
    for (int r = 0; r < 8; ++r) {
        #pragma unroll
        for (int m = 1; m < 16; m <<= 1) {
            q0[r] += __shfl_xor(q0[r], m, 16);
            q1[r] += __shfl_xor(q1[r], m, 16);
        }
    }
    __syncthreads();                       // all LDS ring reads done
    float* part = (float*)&ldsB[0][0];     // part[strip*64 + nh*32 + m]
    if (l15 == 0) {
        const int hb = hi32 >> 2;          // hi*8
        #pragma unroll
        for (int r = 0; r < 8; ++r) {
            part[strip * 64 + nh * 32 + hb + r]      = q0[r];
            part[strip * 64 + nh * 32 + 16 + hb + r] = q1[r];
        }
    }
    __syncthreads();
    if (nh == 0) {                          // 32 lanes -> 32 rows of strip
        float v = part[strip * 64 + lane] + part[strip * 64 + 32 + lane];
        scores[(size_t)b * S_ + sbase + lane] = v;
    }
}

// ---------------------------------------------------------------------------
// Kernel 4: in-place softmax over S per batch row
// ---------------------------------------------------------------------------
__global__ __launch_bounds__(256) void softmax_kernel(float* __restrict__ attn) {
    const int b = blockIdx.x;
    const int t = threadIdx.x;
    __shared__ float red[256];
    float* row = attn + (size_t)b * S_;

    float v[16];
    float mx = -INFINITY;
    #pragma unroll
    for (int i = 0; i < 16; ++i) { v[i] = row[t + i * 256]; mx = fmaxf(mx, v[i]); }
    red[t] = mx; __syncthreads();
    for (int s = 128; s > 0; s >>= 1) {
        if (t < s) red[t] = fmaxf(red[t], red[t + s]);
        __syncthreads();
    }
    mx = red[0]; __syncthreads();

    float sum = 0.0f;
    #pragma unroll
    for (int i = 0; i < 16; ++i) { v[i] = __expf(v[i] - mx); sum += v[i]; }
    red[t] = sum; __syncthreads();
    for (int s = 128; s > 0; s >>= 1) {
        if (t < s) red[t] += red[t + s];
        __syncthreads();
    }
    const float inv = 1.0f / red[0];
    #pragma unroll
    for (int i = 0; i < 16; ++i) row[t + i * 256] = v[i] * inv;
}

// ---------------------------------------------------------------------------
// Kernel 5a: zero context region (atomics accumulate into it)
// ---------------------------------------------------------------------------
__global__ void zero_kernel(float* __restrict__ p) {
    p[blockIdx.x * blockDim.x + threadIdx.x] = 0.0f;
}

// ---------------------------------------------------------------------------
// Kernel 5b: context[b][h] = sum_s attn[b,s] * enc[b,s,h]
// ---------------------------------------------------------------------------
__global__ __launch_bounds__(512) void context_kernel(
    const float* __restrict__ enc,
    const float* __restrict__ attn,
    float* __restrict__ ctx) {
    const int b  = blockIdx.x;
    const int sc = blockIdx.y;        // 0..7
    const int h  = threadIdx.x;       // 0..511
    __shared__ float aw[512];
    const int s0 = sc * 512;
    aw[h] = attn[(size_t)b * S_ + s0 + h];
    __syncthreads();
    float acc = 0.0f;
    const float* base = enc + ((size_t)b * S_ + s0) * H_ + h;
    #pragma unroll 4
    for (int s = 0; s < 512; ++s) acc += aw[s] * base[(size_t)s * H_];
    atomicAdd(&ctx[(size_t)b * H_ + h], acc);
}

// ---------------------------------------------------------------------------
extern "C" void kernel_launch(void* const* d_in, const int* in_sizes, int n_in,
                              void* d_out, int out_size, void* d_ws, size_t ws_size,
                              hipStream_t stream) {
    const float* enc  = (const float*)d_in[0];  // [B,S,H]
    const float* dh   = (const float*)d_in[1];  // [B,H]
    const float* Wenc = (const float*)d_in[2];  // [H,A]
    const float* benc = (const float*)d_in[3];  // [A]
    const float* Wdec = (const float*)d_in[4];  // [H,A]
    const float* bdec = (const float*)d_in[5];  // [A]
    const float* We   = (const float*)d_in[6];  // [A]
    // d_in[7] = b_e : softmax shift-invariant -> unused

    float* out  = (float*)d_out;
    float* ctx  = out;               // [B,H]   = 16384 floats
    float* attn = out + B_ * H_;     // [B,S]   = 131072 floats (also scores)

    char* ws = (char*)d_ws;
    __bf16* WT = (__bf16*)ws;                         // 256 KB
    float*  dp = (float*)(ws + 256 * 1024);           //  32 KB

    wt_convert_kernel<<<(H_ * A_) / 256, 256, 0, stream>>>(Wenc, WT);
    dec_proj_kernel  <<<(B_ * A_) / 256, 256, 0, stream>>>(dh, Wdec, benc, bdec, dp);
    fused_score_kernel<<<(B_ * S_) / 64, 128, 0, stream>>>(enc, WT, dp, We, attn);
    softmax_kernel   <<<B_, 256, 0, stream>>>(attn);
    zero_kernel      <<<(B_ * H_) / 256, 256, 0, stream>>>(ctx);
    context_kernel   <<<dim3(B_, S_ / 512), 512, 0, stream>>>(enc, attn, ctx);
}
